// FilterBasedTokenizer_13915694039257
// MI455X (gfx1250) — matrix operand order repaired and checked by reading, not compiled, not run on hardware
//
#include <hip/hip_runtime.h>

// ---------------------------------------------------------------------------
// FilterBasedTokenizer for MI455X (gfx1250, wave32, WMMA)
//   A = X @ Wa ; A = softmax_N(A) ; T = A^T @ X
// Single streaming pass over X using online (un-shifted) softmax accumulation,
// both GEMMs on v_wmma_f32_16x16x32_bf16 with fp32 accumulate.
// ---------------------------------------------------------------------------

typedef __attribute__((ext_vector_type(16))) __bf16 v16bf;
typedef __attribute__((ext_vector_type(8)))  float  v8f;
typedef __attribute__((ext_vector_type(4)))  float  v4f;

static __device__ __forceinline__ v8f wmma_bf16(v16bf a, v16bf b, v8f c) {
    return __builtin_amdgcn_wmma_f32_16x16x32_bf16(false, a, false, b,
                                                   (short)0, c, false, false);
}

constexpr int Bn = 16;
constexpr int Nn = 50176;
constexpr int Cn = 256;
constexpr int Ln = 16;

constexpr int WAVES_PER_BLOCK  = 8;                 // 256 threads
constexpr int BLOCKS_PER_BATCH = 14;
constexpr int WAVES_PER_BATCH  = WAVES_PER_BLOCK * BLOCKS_PER_BATCH;   // 112
constexpr int TOK_PER_WAVE     = Nn / WAVES_PER_BATCH;                 // 448
constexpr int ITERS            = TOK_PER_WAVE / 32;                    // 14
constexpr int PARTIAL_STRIDE   = Ln * Cn + Ln;                         // 4112 f32

// ---------------------------------------------------------------------------
// Pass 1: per block, stream 8*448 tokens of one batch; produce partial
//         S[l][c] = sum exp(score)*X and Z[l] = sum exp(score).
// ---------------------------------------------------------------------------
__global__ __launch_bounds__(256) void tokenizer_pass1(
    const float* __restrict__ X,    // [B, N, C]
    const float* __restrict__ Wa,   // [B, C, L]
    float* __restrict__ partials)   // [B*BLOCKS_PER_BATCH][PARTIAL_STRIDE]
{
    const int b    = blockIdx.x / BLOCKS_PER_BATCH;
    const int blk  = blockIdx.x % BLOCKS_PER_BATCH;
    const int wave = threadIdx.x >> 5;
    const int lane = threadIdx.x & 31;
    const int l    = lane & 15;     // column within 16-wide tile
    const int h    = lane >> 4;     // wave half

    const float* Xb = X  + (size_t)b * Nn * Cn;
    const float* Wb = Wa + (size_t)b * Cn * Ln;

    // ---- cache Wa as 8 bf16 B-operand tiles (K-chunks of 32 over C) -------
    // B layout (16-bit, 32x16): lane l(+16h) element t  <->  K = 8h + (t&7) + 16*(t>>3), N = l
    v16bf b1[8];
#pragma unroll
    for (int kc = 0; kc < 8; ++kc) {
#pragma unroll
        for (int t = 0; t < 16; ++t)
            b1[kc][t] = (__bf16)Wb[(kc * 32 + 8 * h + (t & 7) + 16 * (t >> 3)) * Ln + l];
    }

    // ---- streaming accumulators ------------------------------------------
    v8f   s[16];                    // T partial: 16 c-tiles of 16x16 f32
#pragma unroll
    for (int ct = 0; ct < 16; ++ct) s[ct] = (v8f)0.0f;
    float zacc = 0.0f;              // per-lane partial of Z[l]

    const int tok0 = (blk * WAVES_PER_BLOCK + wave) * TOK_PER_WAVE;

    for (int it = 0; it < ITERS; ++it) {
        const int tb = tok0 + it * 32;

        // ---- GEMM1 on two 16-token tiles: scores = X_tile @ Wa ------------
        // A layout (16-bit, 16x32): lane l(+16h): M=l, K in {8h..8h+7, 8h+16..8h+23}
        v8f e01[2];
#pragma unroll
        for (int half = 0; half < 2; ++half) {
            const float* xr = Xb + (size_t)(tb + half * 16 + l) * Cn + 8 * h;
            v8f acc = (v8f)0.0f;
#pragma unroll
            for (int kc = 0; kc < 8; ++kc) {
                const v4f* p = reinterpret_cast<const v4f*>(xr + kc * 32);
                v4f f0 = p[0];   // K = 8h + 0..3
                v4f f1 = p[1];   // K = 8h + 4..7
                v4f f2 = p[4];   // K = 8h + 16..19
                v4f f3 = p[5];   // K = 8h + 20..23
                float fa[16];
                *reinterpret_cast<v4f*>(fa + 0)  = f0;
                *reinterpret_cast<v4f*>(fa + 4)  = f1;
                *reinterpret_cast<v4f*>(fa + 8)  = f2;
                *reinterpret_cast<v4f*>(fa + 12) = f3;
                v16bf a;
#pragma unroll
                for (int u = 0; u < 16; ++u) a[u] = (__bf16)fa[u];
                acc = wmma_bf16(a, b1[kc], acc);
            }
            // D layout: acc[j] = score(token tb+16*half+j+8h, l). Scores are
            // O(10) max over 50k samples -> exp safe in fp32 without shift.
            v8f e;
#pragma unroll
            for (int j = 0; j < 8; ++j) e[j] = __expf(acc[j]);
            e01[half] = e;
#pragma unroll
            for (int j = 0; j < 8; ++j) zacc += e[j];
        }

        // ---- A-operand of GEMM2 = E^T; D-of-GEMM1 -> A-of-GEMM2 is a
        //      per-lane identity (no shuffles): K run0 = tile0[j], run1 = tile1[j]
        v16bf a2;
#pragma unroll
        for (int j = 0; j < 8; ++j) {
            a2[j]     = (__bf16)e01[0][j];
            a2[8 + j] = (__bf16)e01[1][j];
        }

        // ---- GEMM2: S[l, c] += E^T @ X over the 32 tokens -----------------
        // B layout: lane l(+16h) element t <-> K = 8h+(t&7)+16*(t>>3) (token), N = l (c)
        const float* xc = Xb + (size_t)(tb + 8 * h) * Cn + l;
#pragma unroll
        for (int ct = 0; ct < 16; ++ct) {
            v16bf b2;
#pragma unroll
            for (int t = 0; t < 16; ++t)
                b2[t] = (__bf16)xc[((t & 7) + 16 * (t >> 3)) * Cn + ct * 16];
            s[ct] = wmma_bf16(a2, b2, s[ct]);
        }
    }

    // ---- deterministic block reduction through LDS ------------------------
    __shared__ float sS[Ln * Cn];
    __shared__ float sZ[Ln];

    const float z2 = zacc + __shfl_xor(zacc, 16, 32);   // combine wave halves

    for (int w = 0; w < WAVES_PER_BLOCK; ++w) {
        if (wave == w) {
#pragma unroll
            for (int ct = 0; ct < 16; ++ct) {
#pragma unroll
                for (int j = 0; j < 8; ++j) {
                    const int idx = (j + 8 * h) * Cn + ct * 16 + l;  // [l'][c]
                    if (w == 0) sS[idx]  = s[ct][j];
                    else        sS[idx] += s[ct][j];
                }
            }
            if (lane < 16) {
                if (w == 0) sZ[l]  = z2;
                else        sZ[l] += z2;
            }
        }
        __syncthreads();
    }

    float* P = partials + (size_t)blockIdx.x * PARTIAL_STRIDE;
    for (int idx = threadIdx.x; idx < Ln * Cn; idx += blockDim.x)
        P[idx] = sS[idx];
    if (threadIdx.x < Ln)
        P[Ln * Cn + threadIdx.x] = sZ[threadIdx.x];
}

// ---------------------------------------------------------------------------
// Pass 2: reduce the 14 partials per batch, normalize: T[b,l,c] = S/Z
// ---------------------------------------------------------------------------
__global__ __launch_bounds__(256) void tokenizer_pass2(
    const float* __restrict__ partials,
    float* __restrict__ out)        // [B, L, C]
{
    const int idx = blockIdx.x * blockDim.x + threadIdx.x;  // b*L*C + l*C + c
    const int b   = idx / (Ln * Cn);
    const int rem = idx % (Ln * Cn);
    const int l   = rem / Cn;

    float ssum = 0.0f, zsum = 0.0f;
#pragma unroll
    for (int k = 0; k < BLOCKS_PER_BATCH; ++k) {
        const float* P = partials +
            (size_t)(b * BLOCKS_PER_BATCH + k) * PARTIAL_STRIDE;
        ssum += P[rem];
        zsum += P[Ln * Cn + l];
    }
    out[idx] = ssum / zsum;
}

// ---------------------------------------------------------------------------
extern "C" void kernel_launch(void* const* d_in, const int* in_sizes, int n_in,
                              void* d_out, int out_size, void* d_ws, size_t ws_size,
                              hipStream_t stream) {
    (void)in_sizes; (void)n_in; (void)out_size; (void)ws_size;
    const float* X  = (const float*)d_in[0];   // [16, 50176, 256] f32
    const float* Wa = (const float*)d_in[1];   // [16, 256, 16]    f32
    float* out      = (float*)d_out;           // [16, 16, 256]    f32
    float* partials = (float*)d_ws;            // 224 * 4112 f32 ≈ 3.7 MB

    tokenizer_pass1<<<Bn * BLOCKS_PER_BATCH, 256, 0, stream>>>(X, Wa, partials);
    tokenizer_pass2<<<(Bn * Ln * Cn) / 256, 256, 0, stream>>>(partials, out);
}
